// DenseValueWindowedPartialLM_34368328303138
// MI455X (gfx1250) — compile-verified, matrix operand order checked by reading, not run on hardware
//
#include <hip/hip_runtime.h>
#include <hip/hip_bf16.h>
#include <math.h>
#include <stdint.h>

typedef __attribute__((ext_vector_type(16))) __bf16 v16bf;
typedef __attribute__((ext_vector_type(8)))  float  v8f;

typedef __attribute__((ext_vector_type(4))) unsigned int tdm_u32x4;
typedef __attribute__((ext_vector_type(8))) int          tdm_i32x8;
typedef __attribute__((ext_vector_type(4))) int          tdm_i32x4;

#if __has_builtin(__builtin_amdgcn_tensor_load_to_lds) && __has_builtin(__builtin_amdgcn_s_wait_tensorcnt)
#define USE_TDM 1
#else
#define USE_TDM 0
#endif

#define kB  2
#define kS  2048
#define kV  32000
#define kE  512
#define kH  1024
#define kMD 256
#define kP  4096
#define kW  128
#define kM  (kB * kS)   // 4096 token rows
#define KC  256         // GEMM k-chunk staged in LDS (all K here are multiples)

// ---------------------------------------------------------------- helpers

__device__ __forceinline__ __bf16 f2bf(float f) {
  __hip_bfloat16 t = __float2bfloat16(f);
  return *reinterpret_cast<__bf16*>(&t);
}

union FragU { v16bf v; float4 q[2]; };

// Load one 16x32 bf16 WMMA fragment (A from row-major [M,K], or B taken as
// a column block of Bᵀ, i.e. a row block of a row-major [N,K] weight).
// Per-ISA layout: lane<16 holds K{0..7,16..23}, lane>=16 holds K{8..15,24..31}.
__device__ __forceinline__ v16bf frag_ld(const __bf16* p, int ld) {
  const int lane = threadIdx.x & 31;
  const int r    = lane & 15;
  const int sel  = lane >> 4;
  const __bf16* rp = p + (size_t)r * ld + sel * 8;
  FragU u;
  u.q[0] = *(const float4*)(rp);
  u.q[1] = *(const float4*)(rp + 16);
  return u.v;
}

__device__ __forceinline__ float sigmoidf_(float x) { return 1.f / (1.f + __expf(-x)); }

// Issue one TDM load of a [rows x KC] bf16 tile (row stride = ld elements)
// from global `src` into LDS `dst`. D# per CDNA5 ISA §8 (2-D tensor, groups
// 2/3 disabled/zero). Caller is responsible for s_wait_tensorcnt + barrier.
#if USE_TDM
__device__ __forceinline__ void tdm_load_tile(const __bf16* src, __bf16* lds_dst,
                                              int rows, int ld) {
  const uint64_t ga  = (uint64_t)(uintptr_t)src;
  const uint32_t lds = (uint32_t)(uintptr_t)(void*)lds_dst;  // low 32b of flat = LDS byte addr
  tdm_u32x4 g0;
  g0[0] = 1u;                                    // count=1 (valid), user mode, no gather
  g0[1] = lds;                                   // lds_addr
  g0[2] = (uint32_t)ga;                          // global_addr[31:0]
  g0[3] = (uint32_t)((ga >> 32) & 0x01FFFFFFu)   // global_addr[56:32]
          | (2u << 30);                          // type=2 ("image")
  tdm_i32x8 g1;
  g1[0] = (1 << 16);                             // workgroup_mask=0, data_size=1 (2 bytes)
  g1[1] = (int)((unsigned)KC << 16);             // tensor_dim0[15:0] in bits[63:48]
  g1[2] = (int)((unsigned)rows << 16);           // tensor_dim0[31:16]=0 | tensor_dim1[15:0]
  g1[3] = (int)((unsigned)KC << 16);             // tensor_dim1[31:16]=0 | tile_dim0=KC
  g1[4] = rows;                                  // tile_dim1=rows, tile_dim2=0
  g1[5] = ld;                                    // tensor_dim0_stride[31:0]
  g1[6] = 0;                                     // stride hi, tensor_dim1_stride lo
  g1[7] = 0;
  tdm_i32x4 z4 = {0, 0, 0, 0};
  tdm_i32x8 z8 = {0, 0, 0, 0, 0, 0, 0, 0};
  // 6-arg form (therock-10.0 headers / clang-23 lane): groups 2,3 disabled.
  __builtin_amdgcn_tensor_load_to_lds(g0, g1, z4, z4, z8, 0);
}
#endif

// ---------------------------------------------------------------- conversions

__global__ void f32_to_bf16_k(const float* __restrict__ src, __bf16* __restrict__ dst, int n) {
  int i = blockIdx.x * 256 + threadIdx.x;
  if (i < n) dst[i] = f2bf(src[i]);
}

__global__ void gather_embed_k(const int* __restrict__ ids, const float* __restrict__ emb,
                               __bf16* __restrict__ xbf) {
  const int row = blockIdx.x;                 // 0..4095 (b*S+s)
  const int id  = ids[row];
  const float* er = emb + (size_t)id * kE;
  __bf16* xr = xbf + (size_t)row * kE;
  for (int e = threadIdx.x; e < kE; e += blockDim.x) xr[e] = f2bf(er[e]);
}

// ---------------------------------------------------------------- WMMA GEMM
// C[M,N] = A[M,K] (bf16, row-major) * W[N,K]^T (bf16, row-major) + rowscale*bias
// Block: 128 threads = 4 waves. Tile 64(M) x 64(N); wave w owns 16 N-columns.
// A staged in LDS in 64xKC chunks by the Tensor Data Mover (TENSORcnt), B
// fragments stream from L2. ACT: 0=none, 1=relu^2.
// SCATTER: atomicAdd into Cscat[row*ldscat + sidx[col]].

template<int ACT, bool BF16OUT, bool SCATTER>
__global__ __launch_bounds__(128)
void wmma_gemm_nt(const __bf16* __restrict__ A, const __bf16* __restrict__ Wt,
                  const float* __restrict__ bias, const float* __restrict__ rowscale,
                  float* __restrict__ Cf, __bf16* __restrict__ Cbf,
                  const int* __restrict__ sidx, float* __restrict__ Cscat, int ldscat,
                  int M, int N, int K) {
  const int m0   = blockIdx.y * 64;
  const int n0   = blockIdx.x * 64;
  const int tid  = threadIdx.x;
  const int wave = tid >> 5;
  const int lane = tid & 31;
  const int wn0  = n0 + wave * 16;

  __shared__ __align__(16) __bf16 As[64 * KC];   // 32 KB
  v8f acc[4] = {};

  for (int k0 = 0; k0 < K; k0 += KC) {
#if USE_TDM
    if (wave == 0) {
      tdm_load_tile(A + (size_t)m0 * K + k0, As, 64, K);
      __builtin_amdgcn_s_wait_tensorcnt(0);
    }
#else
    // manual staging fallback: 2048 float4 chunks, 16 per thread
    for (int i = tid; i < 2048; i += 128) {
      const int r = i >> 5, c = i & 31;
      ((float4*)(As + r * KC))[c] =
          *(const float4*)(A + (size_t)(m0 + r) * K + k0 + c * 8);
    }
#endif
    __syncthreads();

    if (k0 + KC < K)
      __builtin_prefetch(Wt + (size_t)wn0 * K + k0 + KC + (lane & 15) * K, 0, 3);

#pragma unroll
    for (int kk = 0; kk < KC; kk += 32) {
      v16bf bfrag = frag_ld(Wt + (size_t)wn0 * K + k0 + kk, K);
#pragma unroll
      for (int mt = 0; mt < 4; ++mt) {
        v16bf afrag = frag_ld(As + mt * 16 * KC + kk, KC);
        acc[mt] = __builtin_amdgcn_wmma_f32_16x16x32_bf16(
            false, afrag, false, bfrag, (short)0, acc[mt], false, false);
      }
    }
    __syncthreads();
  }

  const int col = wn0 + (lane & 15);
  const int sel = lane >> 4;
  const float bcol = bias[col];
#pragma unroll
  for (int mt = 0; mt < 4; ++mt) {
#pragma unroll
    for (int r = 0; r < 8; ++r) {
      const int row = m0 + mt * 16 + sel * 8 + r;
      const float rs = rowscale ? rowscale[row] : 1.f;
      float v = acc[mt][r] + rs * bcol;
      if (ACT == 1) { v = fmaxf(v, 0.f); v = v * v; }
      if (SCATTER) {
        atomicAdd(Cscat + (size_t)row * ldscat + sidx[col], v);
      } else {
        if (Cf)      Cf [(size_t)row * N + col] = v;
        if (BF16OUT) Cbf[(size_t)row * N + col] = f2bf(v);
      }
    }
  }
}

// ---------------------------------------------------------------- GRU scan
// Persistent kernel: 48 blocks x 128 threads = 6144 threads = one thread per
// (batch, gate-row) output of hg = w_hh @ h. w_hh (12.6 MB) stays L2-resident.
// Device-wide barrier: monotonic atomic counter + s_sleep spin.

#define SCAN_BLOCKS 48

__device__ __forceinline__ void grid_barrier(unsigned* cnt, unsigned& phase) {
  __syncthreads();
  if (threadIdx.x == 0) {
    __threadfence();
    ++phase;
    __hip_atomic_fetch_add(cnt, 1u, __ATOMIC_ACQ_REL, __HIP_MEMORY_SCOPE_AGENT);
    while (__hip_atomic_load(cnt, __ATOMIC_ACQUIRE, __HIP_MEMORY_SCOPE_AGENT) <
           (unsigned)SCAN_BLOCKS * phase) {
      __builtin_amdgcn_s_sleep(2);
    }
  }
  __syncthreads();
}

__global__ __launch_bounds__(128)
void gru_scan_k(const float* __restrict__ gi, const float* __restrict__ w_hh,
                const float* __restrict__ b_hh, float* __restrict__ h,
                float* __restrict__ hg, float* __restrict__ states,
                __bf16* __restrict__ states_bf, unsigned* __restrict__ barcnt) {
  const int tid  = threadIdx.x;
  const int gidx = blockIdx.x * 128 + tid;      // 0..6143
  const int b    = gidx / (3 * kH);             // uniform per block (24 blocks/batch)
  const int g    = gidx % (3 * kH);
  unsigned phase = 0;

  __shared__ __align__(16) float hs[kH];

  // h := 0
  if (gidx < kB * kH) h[gidx] = 0.f;
  grid_barrier(barcnt, phase);

  for (int t = 0; t < kS; ++t) {
    for (int i = tid; i < kH; i += 128) hs[i] = h[b * kH + i];
    __syncthreads();

    // hg[g] = dot(w_hh[g,:], h[b,:]) + b_hh[g]
    const float* wr = w_hh + (size_t)g * kH;
    float acc = 0.f;
#pragma unroll 4
    for (int i = 0; i < kH; i += 4) {
      float4 wv = *(const float4*)(wr + i);
      float4 hv = *(const float4*)(hs + i);
      acc = fmaf(wv.x, hv.x, acc);
      acc = fmaf(wv.y, hv.y, acc);
      acc = fmaf(wv.z, hv.z, acc);
      acc = fmaf(wv.w, hv.w, acc);
    }
    hg[gidx] = acc + b_hh[g];
    grid_barrier(barcnt, phase);

    // combine gates: first 2048 threads finalize h
    if (gidx < kB * kH) {
      const int bb = gidx >> 10, j = gidx & (kH - 1);
      const float* gib = gi + ((size_t)bb * kS + t) * (3 * kH);
      const float* hgb = hg + (size_t)bb * (3 * kH);
      float r = sigmoidf_(gib[j]            + hgb[j]);
      float z = sigmoidf_(gib[kH + j]       + hgb[kH + j]);
      float n = tanhf(    gib[2 * kH + j]   + r * hgb[2 * kH + j]);
      float hn = (1.f - z) * n + z * h[bb * kH + j];
      h[bb * kH + j] = hn;
      size_t so = ((size_t)bb * kS + t) * kH + j;
      states[so]    = hn;
      states_bf[so] = f2bf(hn);
    }
    grid_barrier(barcnt, phase);
  }
}

// ---------------------------------------------------------------- gate

__global__ __launch_bounds__(256)
void gate_k(const float* __restrict__ states, const float* __restrict__ wg,
            const float* __restrict__ bg, const float* __restrict__ mem_scale,
            float* __restrict__ gate, float* __restrict__ rowscale) {
  const int row  = blockIdx.x * 8 + (threadIdx.x >> 5);
  const int lane = threadIdx.x & 31;
  const float* sr = states + (size_t)row * kH;
  float acc = 0.f;
  for (int i = lane; i < kH; i += 32) acc = fmaf(sr[i], wg[i], acc);
#pragma unroll
  for (int o = 16; o; o >>= 1) acc += __shfl_down(acc, o, 32);
  if (lane == 0) {
    float gs = sigmoidf_(acc + bg[0]);
    gate[row]     = gs;
    rowscale[row] = 1.f + gs * mem_scale[0];
  }
}

// ---------------------------------------------------------------- attention
// Block = 128 threads (4 waves) handles 16 queries of one batch.
// Scores via WMMA Q(16xMD) * K^T, masked softmax in LDS, ctx via f32 FMA.

__global__ __launch_bounds__(128)
void attn_k(const __bf16* __restrict__ qb, const __bf16* __restrict__ kb,
            const float* __restrict__ v, float* __restrict__ ctx) {
  const int b    = blockIdx.y;
  const int q0   = blockIdx.x * 16;
  const int tid  = threadIdx.x;
  const int wave = tid >> 5;
  const int lane = tid & 31;
  const int rowbase = b * kS;

  __shared__ float sc[16][160];
  const int kb0    = (q0 >= kW) ? (q0 - kW) : 0;
  const int ntiles = (q0 - kb0) / 16 + 1;              // 1..9

  for (int i = tid; i < 16 * 160; i += 128) ((float*)sc)[i] = -1e30f;
  __syncthreads();

  // Q fragments resident per wave (MD=256 -> 8 k-steps)
  v16bf qf[8];
  const __bf16* qrow0 = qb + (size_t)(rowbase + q0) * kMD;
#pragma unroll
  for (int kk = 0; kk < 8; ++kk) qf[kk] = frag_ld(qrow0 + kk * 32, kMD);

  for (int ti = wave; ti < ntiles; ti += 4) {
    const int kt = kb0 + ti * 16;
    v8f acc = {};
    const __bf16* krow0 = kb + (size_t)(rowbase + kt) * kMD;
#pragma unroll
    for (int kk = 0; kk < 8; ++kk) {
      v16bf kf = frag_ld(krow0 + kk * 32, kMD);
      acc = __builtin_amdgcn_wmma_f32_16x16x32_bf16(
          false, qf[kk], false, kf, (short)0, acc, false, false);
    }
    const int sel = lane >> 4, coll = lane & 15;
    const int kabs = kt + coll;
#pragma unroll
    for (int r = 0; r < 8; ++r) {
      const int qabs = q0 + r + sel * 8;
      const bool valid = (kabs < qabs) && (kabs >= qabs - kW);
      sc[r + sel * 8][ti * 16 + coll] = valid ? acc[r] * (1.f / 16.f) : -1e30f;
    }
  }
  __syncthreads();

  // masked softmax: 8 threads per query row
  const int row = tid >> 3, g = tid & 7;
  const int nk = ntiles * 16;
  float m = -1e30f;
  for (int c = g; c < nk; c += 8) m = fmaxf(m, sc[row][c]);
#pragma unroll
  for (int o = 4; o; o >>= 1) m = fmaxf(m, __shfl_xor(m, o, 8));
  float sum = 0.f;
  for (int c = g; c < nk; c += 8) {
    float s = sc[row][c];
    float e = (s > -1e29f) ? __expf(s - m) : 0.f;   // fully-masked rows -> 0
    sc[row][c] = e;
    sum += e;
  }
#pragma unroll
  for (int o = 4; o; o >>= 1) sum += __shfl_xor(sum, o, 8);
  const float inv = 1.f / fmaxf(sum, 1e-6f);
  __syncthreads();

  // ctx[row, :] : thread g owns 64 of E=512 columns
  float4 cacc[16];
#pragma unroll
  for (int i = 0; i < 16; ++i) cacc[i] = make_float4(0.f, 0.f, 0.f, 0.f);
  const float* vb = v + ((size_t)(rowbase + kb0)) * kE + g * 64;
  for (int c = 0; c < nk; ++c) {
    const float p = sc[row][c] * inv;
    if (p != 0.f) {
      const float4* vr = (const float4*)(vb + (size_t)c * kE);
#pragma unroll
      for (int i = 0; i < 16; ++i) {
        float4 vv = vr[i];
        cacc[i].x = fmaf(p, vv.x, cacc[i].x);
        cacc[i].y = fmaf(p, vv.y, cacc[i].y);
        cacc[i].z = fmaf(p, vv.z, cacc[i].z);
        cacc[i].w = fmaf(p, vv.w, cacc[i].w);
      }
    }
  }
  float4* co = (float4*)(ctx + (size_t)(rowbase + q0 + row) * kE + g * 64);
#pragma unroll
  for (int i = 0; i < 16; ++i) co[i] = cacc[i];
}

// ---------------------------------------------------------------- combo
// combo = bf16(base_feat + gate*mem_scale*ctx); feeds the scatter GEMM.

__global__ void combo_k(const float* __restrict__ base_feat, const float* __restrict__ ctx,
                        const float* __restrict__ gate, const float* __restrict__ mem_scale,
                        __bf16* __restrict__ combo_bf) {
  const size_t i = (size_t)blockIdx.x * 256 + threadIdx.x;
  const int row = (int)(i >> 9);               // /kE
  combo_bf[i] = f2bf(base_feat[i] + gate[row] * mem_scale[0] * ctx[i]);
}

// ---------------------------------------------------------------- launch

extern "C" void kernel_launch(void* const* d_in, const int* in_sizes, int n_in,
                              void* d_out, int out_size, void* d_ws, size_t ws_size,
                              hipStream_t stream) {
  const int*   input_ids = (const int*)  d_in[0];
  const int*   untied    = (const int*)  d_in[1];
  const float* emb       = (const float*)d_in[2];
  const float* w_ih      = (const float*)d_in[3];
  const float* w_hh      = (const float*)d_in[4];
  const float* b_ih      = (const float*)d_in[5];
  const float* b_hh      = (const float*)d_in[6];
  const float* wq        = (const float*)d_in[7];
  const float* bq        = (const float*)d_in[8];
  const float* wk        = (const float*)d_in[9];
  const float* bk        = (const float*)d_in[10];
  const float* wv        = (const float*)d_in[11];
  const float* bv        = (const float*)d_in[12];
  const float* wg        = (const float*)d_in[13];
  const float* bg        = (const float*)d_in[14];
  const float* w_fc      = (const float*)d_in[15];
  const float* b_fc      = (const float*)d_in[16];
  const float* w_hp      = (const float*)d_in[17];
  const float* b_hp      = (const float*)d_in[18];
  const float* out_bias  = (const float*)d_in[19];
  const float* w_ph      = (const float*)d_in[20];
  const float* b_ph      = (const float*)d_in[21];
  const float* mem_scale = (const float*)d_in[22];
  float* out = (float*)d_out;

  char* wsp = (char*)d_ws;
  size_t off = 0;
  auto alloc = [&](size_t bytes) -> void* {
    void* p = wsp + off;
    off += (bytes + 255) & ~(size_t)255;
    return p;
  };

  __bf16* emb_bf   = (__bf16*)alloc((size_t)kV * kE * 2);
  __bf16* wih_bf   = (__bf16*)alloc((size_t)3 * kH * kE * 2);
  __bf16* wq_bf    = (__bf16*)alloc((size_t)kMD * kH * 2);
  __bf16* wk_bf    = (__bf16*)alloc((size_t)kMD * kH * 2);
  __bf16* wv_bf    = (__bf16*)alloc((size_t)kE * kH * 2);
  __bf16* wfc_bf   = (__bf16*)alloc((size_t)4 * kE * kH * 2);
  __bf16* whp_bf   = (__bf16*)alloc((size_t)kE * 4 * kE * 2);
  __bf16* wph_bf   = (__bf16*)alloc((size_t)kP * kE * 2);
  __bf16* x_bf     = (__bf16*)alloc((size_t)kM * kE * 2);
  float*  gi       = (float*) alloc((size_t)kM * 3 * kH * 4);
  float*  h        = (float*) alloc((size_t)kB * kH * 4);
  float*  hg       = (float*) alloc((size_t)kB * 3 * kH * 4);
  unsigned* barcnt = (unsigned*)alloc(256);
  float*  states   = (float*) alloc((size_t)kM * kH * 4);
  __bf16* states_bf= (__bf16*)alloc((size_t)kM * kH * 2);
  __bf16* hf_bf    = (__bf16*)alloc((size_t)kM * 4 * kE * 2);
  float*  base_feat= (float*) alloc((size_t)kM * kE * 4);
  __bf16* bfeat_bf = (__bf16*)alloc((size_t)kM * kE * 2);
  __bf16* q_bf     = (__bf16*)alloc((size_t)kM * kMD * 2);
  __bf16* k_bf     = (__bf16*)alloc((size_t)kM * kMD * 2);
  float*  v_f      = (float*) alloc((size_t)kM * kE * 4);
  float*  ctx      = (float*) alloc((size_t)kM * kE * 4);
  float*  gate     = (float*) alloc((size_t)kM * 4);
  float*  rowscale = (float*) alloc((size_t)kM * 4);
  __bf16* combo_bf = (__bf16*)alloc((size_t)kM * kE * 2);

  (void)hipMemsetAsync(barcnt, 0, 256, stream);

  // weight conversions f32 -> bf16
  auto conv = [&](const float* s, __bf16* d, int n) {
    f32_to_bf16_k<<<(n + 255) / 256, 256, 0, stream>>>(s, d, n);
  };
  conv(emb,  emb_bf, kV * kE);
  conv(w_ih, wih_bf, 3 * kH * kE);
  conv(wq,   wq_bf,  kMD * kH);
  conv(wk,   wk_bf,  kMD * kH);
  conv(wv,   wv_bf,  kE * kH);
  conv(w_fc, wfc_bf, 4 * kE * kH);
  conv(w_hp, whp_bf, kE * 4 * kE);
  conv(w_ph, wph_bf, kP * kE);

  // embedding gather (bf16)
  gather_embed_k<<<kM, 128, 0, stream>>>(input_ids, emb, x_bf);

  // gi = x @ w_ih^T + b_ih          [4096 x 3072]
  wmma_gemm_nt<0, false, false><<<dim3(3 * kH / 64, kM / 64), 128, 0, stream>>>(
      x_bf, wih_bf, b_ih, nullptr, gi, nullptr, nullptr, nullptr, 0, kM, 3 * kH, kE);

  // sequential GRU scan (persistent, device-wide barriers)
  gru_scan_k<<<SCAN_BLOCKS, 128, 0, stream>>>(gi, w_hh, b_hh, h, hg, states,
                                              states_bf, barcnt);

  // hf = relu(states @ w_fc^T + b_fc)^2   [4096 x 2048] (bf16 only)
  wmma_gemm_nt<1, true, false><<<dim3(4 * kE / 64, kM / 64), 128, 0, stream>>>(
      states_bf, wfc_bf, b_fc, nullptr, nullptr, hf_bf, nullptr, nullptr, 0,
      kM, 4 * kE, kH);

  // base_feat = hf @ w_hp^T + b_hp        [4096 x 512] (f32 + bf16)
  wmma_gemm_nt<0, true, false><<<dim3(kE / 64, kM / 64), 128, 0, stream>>>(
      hf_bf, whp_bf, b_hp, nullptr, base_feat, bfeat_bf, nullptr, nullptr, 0,
      kM, kE, 4 * kE);

  // q, k (bf16), v (f32)
  wmma_gemm_nt<0, true, false><<<dim3(kMD / 64, kM / 64), 128, 0, stream>>>(
      states_bf, wq_bf, bq, nullptr, nullptr, q_bf, nullptr, nullptr, 0, kM, kMD, kH);
  wmma_gemm_nt<0, true, false><<<dim3(kMD / 64, kM / 64), 128, 0, stream>>>(
      states_bf, wk_bf, bk, nullptr, nullptr, k_bf, nullptr, nullptr, 0, kM, kMD, kH);
  wmma_gemm_nt<0, false, false><<<dim3(kE / 64, kM / 64), 128, 0, stream>>>(
      states_bf, wv_bf, bv, nullptr, v_f, nullptr, nullptr, nullptr, 0, kM, kE, kH);

  // gate + rowscale (= 1 + gate*mem_scale)
  gate_k<<<kM / 8, 256, 0, stream>>>(states, wg, bg, mem_scale, gate, rowscale);

  // windowed attention -> ctx
  attn_k<<<dim3(kS / 16, kB), 128, 0, stream>>>(q_bf, k_bf, v_f, ctx);

  // base_logits = base_feat @ emb^T + out_bias -> d_out  [4096 x 32000]
  wmma_gemm_nt<0, false, false><<<dim3(kV / 64, kM / 64), 128, 0, stream>>>(
      bfeat_bf, emb_bf, out_bias, nullptr, out, nullptr, nullptr, nullptr, 0,
      kM, kV, kE);

  // combo = base_feat + gate*mem_scale*ctx (bf16)
  combo_k<<<(kM * kE) / 256, 256, 0, stream>>>(base_feat, ctx, gate, mem_scale, combo_bf);

  // total_partial = combo @ w_ph^T + rowscale*b_ph, scatter-added into d_out
  wmma_gemm_nt<0, false, true><<<dim3(kP / 64, kM / 64), 128, 0, stream>>>(
      combo_bf, wph_bf, b_ph, rowscale, nullptr, nullptr, untied, out, kV,
      kM, kP, kE);
}
